// MTN_b_27986006900834
// MI455X (gfx1250) — compile-verified, matrix operand order checked by reading, not run on hardware
//
#include <hip/hip_runtime.h>
#include <stdint.h>

typedef __bf16 bf16;
typedef __attribute__((ext_vector_type(16))) __bf16 v16bf;
typedef __attribute__((ext_vector_type(8)))  __bf16 v8bf;
typedef __attribute__((ext_vector_type(8)))  float  v8f;
typedef __attribute__((ext_vector_type(4)))  float  v4f;

#define CEILDIV(a,b) (((a)+(b)-1)/(b))

// Model dims (fixed by the reference)
static const int D_   = 200;
static const int NC_  = 250;
// Padded dims
static const int DP   = 208;   // D padded to 16
static const int K1P  = 608;   // 3D padded to 32
static const int K2P  = 416;   // 2D padded to 32
static const int K3P  = 224;   // D padded to 32
static const int NCP  = 256;   // NC padded to 64
static const int GROWS= 832;   // 4 gates * 208
static const int BPAD = 256;   // bias padding (>= grid_n*64 for DP-sized outputs)

__device__ __forceinline__ float sigmoidf_(float x) { return 1.0f / (1.0f + __expf(-x)); }

// ---------------- weight / bias packing ----------------

__global__ void pack_w_k(const float* __restrict__ src, int sM, int sK,
                         bf16* __restrict__ dst, int dM, int dK) {
  int i = blockIdx.x * 256 + threadIdx.x;
  if (i >= dM * dK) return;
  int m = i / dK, k = i % dK;
  float v = (m < sM && k < sK) ? src[m * sK + k] : 0.0f;
  dst[i] = (bf16)v;
}

// Gates weight: rows [0..207]=i(ix|ih), [208..415]=u(ux|uh), [416..623]=o(ox|oh), [624..831]=fx(fx|0)
__global__ void pack_wg_k(const float* __restrict__ ix, const float* __restrict__ ih,
                          const float* __restrict__ ux, const float* __restrict__ uh,
                          const float* __restrict__ ox, const float* __restrict__ oh,
                          const float* __restrict__ fx, bf16* __restrict__ dst) {
  int i = blockIdx.x * 256 + threadIdx.x;
  if (i >= GROWS * K2P) return;
  int r = i / K2P, k = i % K2P;
  int g = r / DP, m = r % DP;
  float v = 0.0f;
  if (m < D_) {
    if (g < 3) {
      const float* wx = (g == 0) ? ix : (g == 1) ? ux : ox;
      const float* wh = (g == 0) ? ih : (g == 1) ? uh : oh;
      if (k < D_)           v = wx[m * D_ + k];
      else if (k < 2 * D_)  v = wh[m * D_ + (k - D_)];
    } else {
      if (k < D_)           v = fx[m * D_ + k];
    }
  }
  dst[i] = (bf16)v;
}

__global__ void pack_bias_k(const float* __restrict__ src, int sN,
                            float* __restrict__ dst, int dN) {
  int i = blockIdx.x * 256 + threadIdx.x;
  if (i >= dN) return;
  dst[i] = (i < sN) ? src[i] : 0.0f;
}

__global__ void pack_bg_k(const float* __restrict__ ixb, const float* __restrict__ ihb,
                          const float* __restrict__ uxb, const float* __restrict__ uhb,
                          const float* __restrict__ oxb, const float* __restrict__ ohb,
                          const float* __restrict__ fxb, float* __restrict__ dst) {
  int i = blockIdx.x * 256 + threadIdx.x;
  if (i >= GROWS) return;
  int g = i / DP, m = i % DP;
  float v = 0.0f;
  if (m < D_) {
    if      (g == 0) v = ixb[m] + ihb[m];
    else if (g == 1) v = uxb[m] + uhb[m];
    else if (g == 2) v = oxb[m] + ohb[m];
    else             v = fxb[m];
  }
  dst[i] = v;
}

// ---------------- activation buffer prep ----------------

// A2[n][0:200] written later by GEMM1 epilogue; here fill [200:400]=child_h[:,3], [400:416]=0
__global__ void prep_A2_k(const float* __restrict__ child_h, bf16* __restrict__ A2, int n_nodes) {
  int i = blockIdx.x * 256 + threadIdx.x;
  if (i >= n_nodes * 216) return;
  int n = i / 216, j = i % 216;
  float v = (j < D_) ? child_h[(size_t)n * 800 + 600 + j] : 0.0f;
  A2[(size_t)n * K2P + 200 + j] = (bf16)v;
}

// A3[n][0:200]=emb[tokens[n]], [400:416]=0 ; [200:400] written later by GEMM2 epilogue
__global__ void prep_A3_k(const int* __restrict__ tok, const float* __restrict__ emb,
                          bf16* __restrict__ A3, int n_nodes) {
  int i = blockIdx.x * 256 + threadIdx.x;
  if (i >= n_nodes * 216) return;
  int n = i / 216, j = i % 216;
  if (j < D_) {
    A3[(size_t)n * K2P + j] = (bf16)emb[(size_t)tok[n] * D_ + j];
  } else {
    A3[(size_t)n * K2P + 400 + (j - D_)] = (bf16)0.0f;
  }
}

// ---------------- WMMA GEMM: C[M x n_limit] = act(A @ B^T + bias) ----------------
// Block tile 128x64, BK=32, 4 waves, each wave = 32x64 (2x4 WMMA 16x16x32 bf16 tiles)
// Register-staged, double-buffered LDS pipeline: 1 barrier per K-slice.
// Out-of-range A/B rows are address-clamped (their C rows/cols are store-masked);
// A's K-tail garbage is harmless because packed B is zero for k >= k_valid.
// Only the LAST K-slice may be partial for AF32 sources -> branchless steady state.
// MODE 0: relu -> bf16 dst (stride dst_stride, col offset dst_coloff)
// MODE 1: gates: g=col/208 : 0,2->sigmoid ; 1->tanh ; 3->raw ; fp32 dst
// MODE 2: f pre-act: + fxpre from G[(row/4)*832 + 624 + col] ; fp32 dst
// MODE 3: +bias only ; fp32 dst (logits)
template <int MODE, bool AF32>
__global__ __launch_bounds__(128) void gemm_k(
    const void* __restrict__ Aptr, int a_stride, int a_kvalid, int M_rows,
    const bf16* __restrict__ Bw, int B_rows,
    const float* __restrict__ bias, int Ksteps,
    void* __restrict__ Dst, int dst_stride, int dst_coloff, int n_limit,
    const float* __restrict__ Gfx) {
  const int BM = 128, BN = 64, BK = 32, LDA = 40;
  __shared__ bf16 As[2][BM * LDA];
  __shared__ bf16 Bs[2][BN * LDA];

  const int t = threadIdx.x;
  const int lane = t & 31;
  const int wave = t >> 5;
  const int bm = blockIdx.y * BM;
  const int bn = blockIdx.x * BN;
  const int Kpad = Ksteps * BK;

  v8f acc[2][4];
#pragma unroll
  for (int a = 0; a < 2; ++a)
#pragma unroll
    for (int b = 0; b < 4; ++b) acc[a][b] = (v8f){0.f,0.f,0.f,0.f,0.f,0.f,0.f,0.f};

  // Clamped staging addresses (no divergent guards; see note above).
  const int arow = min(bm + t, M_rows - 1);
  const int brow = min(bn + (t >> 1), B_rows - 1);
  const int bko = (t & 1) * 16;
  const char* Abase = (const char*)Aptr +
      (AF32 ? (size_t)arow * a_stride * 4 : (size_t)arow * a_stride * 2);
  const bf16* Bbase = Bw + (size_t)brow * Kpad + bko;

  v8bf ar[4];  // staged A row slice (32 bf16)
  v8bf br[2];  // staged B half-row slice (16 bf16)

  auto cvt8 = [](v4f a, v4f b) -> v8bf {
    v8bf r;
    r[0] = (bf16)a.x; r[1] = (bf16)a.y; r[2] = (bf16)a.z; r[3] = (bf16)a.w;
    r[4] = (bf16)b.x; r[5] = (bf16)b.y; r[6] = (bf16)b.z; r[7] = (bf16)b.w;
    return r;
  };

  // Full slice: unguarded 16B vector loads.
  auto stageA_full = [&](int ks) {
    if (AF32) {
      const v4f* src = (const v4f*)((const float*)Abase + ks * BK);
#pragma unroll
      for (int c = 0; c < 4; ++c) ar[c] = cvt8(src[2 * c], src[2 * c + 1]);
    } else {
      const bf16* src = (const bf16*)Abase + ks * BK;
#pragma unroll
      for (int c = 0; c < 4; ++c) ar[c] = *(const v8bf*)(src + c * 8);
    }
  };
  // Tail slice (AF32 only): uniform per-chunk guards, zeros past k_valid.
  auto stageA_tail = [&](int ks) {
    const float* src = (const float*)Abase + ks * BK;
    const int kbase = ks * BK;
    const v4f vz = (v4f){0.f, 0.f, 0.f, 0.f};
#pragma unroll
    for (int c = 0; c < 4; ++c) {
      v4f v0 = (kbase + c * 8 + 4 <= a_kvalid) ? *(const v4f*)(src + c * 8) : vz;
      v4f v1 = (kbase + c * 8 + 8 <= a_kvalid) ? *(const v4f*)(src + c * 8 + 4) : vz;
      ar[c] = cvt8(v0, v1);
    }
  };
  auto stageA = [&](int ks) {
    if (AF32 && (ks + 1) * BK > a_kvalid) stageA_tail(ks);   // uniform branch
    else                                  stageA_full(ks);
  };
  auto stageB = [&](int ks) {
    const bf16* src = Bbase + ks * BK;
    br[0] = *(const v8bf*)(src);
    br[1] = *(const v8bf*)(src + 8);
  };

  stageA(0);
  stageB(0);

  const int half = lane >> 4;
  const int r16 = lane & 15;
  int p = 0;

  for (int ks = 0; ks < Ksteps; ++ks) {
    // commit staged regs -> LDS[p]
    bf16* asr = &As[p][t * LDA];
#pragma unroll
    for (int c = 0; c < 4; ++c) *(v8bf*)(asr + c * 8) = ar[c];
    bf16* bsr = &Bs[p][(t >> 1) * LDA + bko];
    *(v8bf*)(bsr + 0) = br[0];
    *(v8bf*)(bsr + 8) = br[1];
    __syncthreads();

    // prefetch next slice into regs (overlaps with WMMAs below)
    if (ks + 1 < Ksteps) { stageA(ks + 1); stageB(ks + 1); }

    // fragments: lane<16 -> row=lane, K{0..7,16..23}; lane>=16 -> K{8..15,24..31}
    v16bf af[2], bfrag[4];
#pragma unroll
    for (int mt = 0; mt < 2; ++mt) {
      const bf16* q = &As[p][(wave * 32 + mt * 16 + r16) * LDA + half * 8];
      v8bf lo = *(const v8bf*)q;
      v8bf hi = *(const v8bf*)(q + 16);
      af[mt] = __builtin_shufflevector(lo, hi, 0,1,2,3,4,5,6,7,8,9,10,11,12,13,14,15);
    }
#pragma unroll
    for (int nt = 0; nt < 4; ++nt) {
      const bf16* q = &Bs[p][(nt * 16 + r16) * LDA + half * 8];
      v8bf lo = *(const v8bf*)q;
      v8bf hi = *(const v8bf*)(q + 16);
      bfrag[nt] = __builtin_shufflevector(lo, hi, 0,1,2,3,4,5,6,7,8,9,10,11,12,13,14,15);
    }
#pragma unroll
    for (int mt = 0; mt < 2; ++mt)
#pragma unroll
      for (int nt = 0; nt < 4; ++nt)
        acc[mt][nt] = __builtin_amdgcn_wmma_f32_16x16x32_bf16(
            false, af[mt], false, bfrag[nt], (short)0, acc[mt][nt], false, false);

    p ^= 1;
  }

  // ---- epilogue: D layout: row = base + 8*half + i, col = base + (lane&15) ----
#pragma unroll
  for (int mt = 0; mt < 2; ++mt) {
#pragma unroll
    for (int nt = 0; nt < 4; ++nt) {
      const int col = bn + nt * 16 + r16;
      const float bv = bias[col];             // bias arrays padded to grid width
      const int row0 = bm + wave * 32 + mt * 16 + half * 8;
#pragma unroll
      for (int i = 0; i < 8; ++i) {
        const int row = row0 + i;
        if (row < M_rows && col < n_limit) {
          float x = acc[mt][nt][i] + bv;
          if (MODE == 0) {
            x = fmaxf(x, 0.0f);
            ((bf16*)Dst)[(size_t)row * dst_stride + dst_coloff + col] = (bf16)x;
          } else if (MODE == 1) {
            const int g = col / DP;
            float y = (g == 1) ? tanhf(x) : ((g == 3) ? x : sigmoidf_(x));
            ((float*)Dst)[(size_t)row * dst_stride + col] = y;
          } else if (MODE == 2) {
            x += Gfx[(size_t)(row >> 2) * GROWS + 3 * DP + col];
            ((float*)Dst)[(size_t)row * dst_stride + col] = x;
          } else {
            ((float*)Dst)[(size_t)row * dst_stride + col] = x;
          }
        }
      }
    }
  }
}

// ---------------- combine: c = i*u + sum_c sigmoid(fpre)*child_c ; h = o*tanh(c) ----------------
__global__ void combine_k(const float* __restrict__ G, const float* __restrict__ FP,
                          const float* __restrict__ child_c, bf16* __restrict__ H, int n_nodes) {
  int i = blockIdx.x * 256 + threadIdx.x;
  if (i >= n_nodes * K3P) return;
  int n = i / K3P, m = i % K3P;
  if (m < D_) {
    const float* g = G + (size_t)n * GROWS;
    float iv = g[m], uv = g[DP + m], ov = g[2 * DP + m];
    float c = iv * uv;
    const float* fp = FP + (size_t)n * 4 * DP;
    const float* cc = child_c + (size_t)n * 800;
#pragma unroll
    for (int ch = 0; ch < 4; ++ch)
      c += sigmoidf_(fp[ch * DP + m]) * cc[ch * D_ + m];
    H[(size_t)n * K3P + m] = (bf16)(ov * tanhf(c));
  } else {
    H[(size_t)n * K3P + m] = (bf16)0.0f;
  }
}

// ---------------- host ----------------

extern "C" void kernel_launch(void* const* d_in, const int* in_sizes, int n_in,
                              void* d_out, int out_size, void* d_ws, size_t ws_size,
                              hipStream_t stream) {
  const int*   tokens  = (const int*)  d_in[0];
  const float* child_c = (const float*)d_in[1];
  const float* child_h = (const float*)d_in[2];
  const float* emb     = (const float*)d_in[3];
  const float* p1_w = (const float*)d_in[4],  *p1_b = (const float*)d_in[5];
  const float* p2_w = (const float*)d_in[6],  *p2_b = (const float*)d_in[7];
  const float* ix_w = (const float*)d_in[8],  *ix_b = (const float*)d_in[9];
  const float* ih_w = (const float*)d_in[10], *ih_b = (const float*)d_in[11];
  const float* fx_w = (const float*)d_in[12], *fx_b = (const float*)d_in[13];
  const float* fh_w = (const float*)d_in[14], *fh_b = (const float*)d_in[15];
  const float* ux_w = (const float*)d_in[16], *ux_b = (const float*)d_in[17];
  const float* uh_w = (const float*)d_in[18], *uh_b = (const float*)d_in[19];
  const float* ox_w = (const float*)d_in[20], *ox_b = (const float*)d_in[21];
  const float* oh_w = (const float*)d_in[22], *oh_b = (const float*)d_in[23];
  const float* fc2_w = (const float*)d_in[24], *fc2_b = (const float*)d_in[25];

  const int n_nodes = in_sizes[0];  // 60000

  // workspace carve
  char* base = (char*)d_ws;
  size_t cur = 0;
  auto alloc = [&](size_t bytes) -> void* {
    void* p = base + cur;
    cur = (cur + bytes + 255) & ~(size_t)255;
    return p;
  };
  bf16*  Wp1 = (bf16*) alloc((size_t)DP   * K1P * 2);
  bf16*  Wp2 = (bf16*) alloc((size_t)DP   * K2P * 2);
  bf16*  Wg  = (bf16*) alloc((size_t)GROWS* K2P * 2);
  bf16*  Wf  = (bf16*) alloc((size_t)DP   * K3P * 2);
  bf16*  Wc  = (bf16*) alloc((size_t)NCP  * K3P * 2);
  float* b1  = (float*)alloc(BPAD  * 4);
  float* b2  = (float*)alloc(BPAD  * 4);
  float* bg  = (float*)alloc(GROWS * 4);
  float* bfb = (float*)alloc(BPAD  * 4);
  float* bc  = (float*)alloc(BPAD  * 4);
  bf16*  A2  = (bf16*) alloc((size_t)n_nodes * K2P * 2);       // [cat(control, h3) pad]
  bf16*  A3  = (bf16*) alloc((size_t)n_nodes * K2P * 2);       // [cat(x, hsum) pad]
  float* G   = (float*)alloc((size_t)n_nodes * GROWS * 4);     // i,u,o activations + fxpre
  float* FP  = (float*)alloc((size_t)n_nodes * 4 * DP * 4);    // f pre-activations
  bf16*  H   = (bf16*) alloc((size_t)n_nodes * K3P * 2);       // h padded

  const dim3 blk256(256), blk128(128);

  // pack weights & biases
  pack_w_k<<<CEILDIV(DP * K1P, 256), blk256, 0, stream>>>(p1_w, D_, 3 * D_, Wp1, DP, K1P);
  pack_w_k<<<CEILDIV(DP * K2P, 256), blk256, 0, stream>>>(p2_w, D_, 2 * D_, Wp2, DP, K2P);
  pack_wg_k<<<CEILDIV(GROWS * K2P, 256), blk256, 0, stream>>>(ix_w, ih_w, ux_w, uh_w, ox_w, oh_w, fx_w, Wg);
  pack_w_k<<<CEILDIV(DP * K3P, 256), blk256, 0, stream>>>(fh_w, D_, D_, Wf, DP, K3P);
  pack_w_k<<<CEILDIV(NCP * K3P, 256), blk256, 0, stream>>>(fc2_w, NC_, D_, Wc, NCP, K3P);
  pack_bias_k<<<1, blk256, 0, stream>>>(p1_b, D_, b1, BPAD);
  pack_bias_k<<<1, blk256, 0, stream>>>(p2_b, D_, b2, BPAD);
  pack_bg_k<<<CEILDIV(GROWS, 256), blk256, 0, stream>>>(ix_b, ih_b, ux_b, uh_b, ox_b, oh_b, fx_b, bg);
  pack_bias_k<<<1, blk256, 0, stream>>>(fh_b, D_, bfb, BPAD);
  pack_bias_k<<<1, blk256, 0, stream>>>(fc2_b, NC_, bc, BPAD);

  // activation buffer prep
  prep_A2_k<<<CEILDIV(n_nodes * 216, 256), blk256, 0, stream>>>(child_h, A2, n_nodes);
  prep_A3_k<<<CEILDIV(n_nodes * 216, 256), blk256, 0, stream>>>(tokens, emb, A3, n_nodes);

  const int gm  = CEILDIV(n_nodes, 128);
  const int gm4 = CEILDIV(n_nodes * 4, 128);

  // GEMM1: control = relu(child_h[:,0:3] @ p1^T + b1) -> A2[:, 0:200]
  gemm_k<0, true><<<dim3(CEILDIV(DP, 64), gm), blk128, 0, stream>>>(
      child_h, 800, 600, n_nodes, Wp1, DP, b1, K1P / 32, A2, K2P, 0, D_, nullptr);
  // GEMM2: hsum = relu(A2 @ p2^T + b2) -> A3[:, 200:400]
  gemm_k<0, false><<<dim3(CEILDIV(DP, 64), gm), blk128, 0, stream>>>(
      A2, K2P, K2P, n_nodes, Wp2, DP, b2, K2P / 32, A3, K2P, 200, D_, nullptr);
  // GEMM3: gates = act(A3 @ Wg^T + bg) -> G[n][g][m] (i=sig, u=tanh, o=sig, fx=raw)
  gemm_k<1, false><<<dim3(CEILDIV(GROWS, 64), gm), blk128, 0, stream>>>(
      A3, K2P, K2P, n_nodes, Wg, GROWS, bg, K2P / 32, G, GROWS, 0, GROWS, nullptr);
  // GEMM4: fpre = child_h(flat [4N,200]) @ fh^T + fh_b + fxpre -> FP
  gemm_k<2, true><<<dim3(CEILDIV(DP, 64), gm4), blk128, 0, stream>>>(
      child_h, 200, 200, n_nodes * 4, Wf, DP, bfb, K3P / 32, FP, DP, 0, DP, G);
  // combine: h = o * tanh(i*u + sum_c sigmoid(fpre)*child_c) -> H bf16
  combine_k<<<CEILDIV(n_nodes * K3P, 256), blk256, 0, stream>>>(G, FP, child_c, H, n_nodes);
  // GEMM5: logits = H @ fc2^T + fc2_b -> d_out [N, 250] fp32
  gemm_k<3, false><<<dim3(CEILDIV(NCP, 64), gm), blk128, 0, stream>>>(
      H, K3P, K3P, n_nodes, Wc, NCP, bc, K3P / 32, d_out, NC_, 0, NC_, nullptr);
}
